// TransformerEncoder_17497696764619
// MI455X (gfx1250) — matrix operand order repaired and checked by reading, not compiled, not run on hardware
//
#include <hip/hip_runtime.h>

// ---------------------------------------------------------------- constants
#define LAYERS 4
#define NH     16
#define DMODEL 1024
#define HD     64      // DMODEL / NH
#define BATCH  8
#define SLEN   512
#define BS     (BATCH * SLEN)          // 4096 rows
#define DD     (DMODEL * DMODEL)       // 1,048,576
#define BSD    (BS * DMODEL)           // 4,194,304
#define SCORES_PER_LAYER ((size_t)BATCH * NH * SLEN * SLEN)

typedef __attribute__((ext_vector_type(16))) __bf16 v16bf;
typedef __attribute__((ext_vector_type(8)))  float v8f;

// ---------------------------------------------------------------- helpers
__device__ __forceinline__ unsigned short f2bf(float x) {
  unsigned int u = __float_as_uint(x);
  unsigned int r = u + 0x7FFFu + ((u >> 16) & 1u);   // round-to-nearest-even
  return (unsigned short)(r >> 16);
}

__device__ __forceinline__ v16bf load_frag_pair(const unsigned short* p0,
                                                const unsigned short* p1) {
  v16bf r;
  ((uint4*)&r)[0] = *(const uint4*)p0;   // elements 0..7   (16B)
  ((uint4*)&r)[1] = *(const uint4*)p1;   // elements 8..15  (16B)
  return r;
}

// A fragment, 16x32 bf16, A row-major [M x K] (lda = K)
__device__ __forceinline__ v16bf fragA(const unsigned short* A, int lda,
                                       int row0, int kk, int lane) {
  int r = lane & 15, half = lane >> 4;
  const unsigned short* p = A + (size_t)(row0 + r) * lda + kk + half * 8;
  return load_frag_pair(p, p + 16);
}

// B fragment, 32x16 bf16, from B^T stored row-major [N x K] (ldb = K)
__device__ __forceinline__ v16bf fragB(const unsigned short* Bt, int ldb,
                                       int col0, int kk, int lane) {
  int n = lane & 15, half = lane >> 4;
  const unsigned short* p = Bt + (size_t)(col0 + n) * ldb + kk + half * 8;
  return load_frag_pair(p, p + 16);
}

__device__ __forceinline__ v8f wmma_bf16(v16bf a, v16bf b, v8f c) {
  return __builtin_amdgcn_wmma_f32_16x16x32_bf16(false, a, false, b,
                                                 (short)0, c, false, false);
}

// CDNA5 async global->LDS copy (ASYNCcnt path), 16B per lane per issue.
__device__ __forceinline__ void async_b128(unsigned lds_addr,
                                           const unsigned short* g) {
  asm volatile("global_load_async_to_lds_b128 %0, %1, off"
               :: "v"(lds_addr), "v"((unsigned long long)(size_t)g)
               : "memory");
}

#define ACT_NONE 0
#define ACT_SELU 1
#define ACT_RELU 2
__device__ __forceinline__ float apply_act(float x, int act) {
  if (act == ACT_SELU) {
    const float lam = 1.0507009873554805f, alp = 1.6732632423543772f;
    return x > 0.f ? lam * x : lam * alp * (__expf(x) - 1.f);
  }
  if (act == ACT_RELU) return fmaxf(x, 0.f);
  return x;
}

// ---------------------------------------------------------------- conversions
__global__ __launch_bounds__(256) void cvt_bf16_kernel(
    const float* __restrict__ src, unsigned short* __restrict__ dst, int n) {
  int i = blockIdx.x * 256 + threadIdx.x;
  if (i < n) dst[i] = f2bf(src[i]);
}

// W [K x N] f32  ->  Wt [N x K] bf16   (tiled through LDS)
__global__ __launch_bounds__(256) void transpose_cvt_kernel(
    const float* __restrict__ W, unsigned short* __restrict__ Wt, int K, int N) {
  __shared__ unsigned short tile[32][33];
  int n0 = blockIdx.x * 32, k0 = blockIdx.y * 32;
  int tx = threadIdx.x & 31, ty = threadIdx.x >> 5;  // 32 x 8
#pragma unroll
  for (int i = 0; i < 32; i += 8)
    tile[ty + i][tx] = f2bf(W[(size_t)(k0 + ty + i) * N + n0 + tx]);
  __syncthreads();
#pragma unroll
  for (int i = 0; i < 32; i += 8)
    Wt[(size_t)(n0 + ty + i) * K + k0 + tx] = tile[tx][ty + i];
}

// q f32 [B,S,NH,HD] -> qbf bf16 [B,NH,S,HD]
__global__ __launch_bounds__(256) void permute_qk_kernel(
    const float* __restrict__ in, unsigned short* __restrict__ out) {
  int i = blockIdx.x * 256 + threadIdx.x;          // 0 .. 2^22-1
  int d = i & (HD - 1);
  int h = (i >> 6) & (NH - 1);
  int s = (i >> 10) & (SLEN - 1);
  int b = i >> 19;
  out[((((size_t)b * NH + h) * SLEN + s) << 6) + d] = f2bf(in[i]);
}

// h f32 [B,S,NH,HD] -> vt bf16 [B,NH,HD,S]   (V transposed for B-fragments)
__global__ __launch_bounds__(256) void transpose_v_kernel(
    const float* __restrict__ in, unsigned short* __restrict__ out) {
  int i = blockIdx.x * 256 + threadIdx.x;
  int d = i & (HD - 1);
  int h = (i >> 6) & (NH - 1);
  int s = (i >> 10) & (SLEN - 1);
  int b = i >> 19;
  out[(((size_t)b * NH + h) * HD + d) * SLEN + s] = f2bf(in[i]);
}

// ---------------------------------------------------------------- dense GEMM
// C[M,N] = act( A_bf16[M,K] @ Bt_bf16[N,K]^T + bias + resid )
// block = 128 (4 waves), block tile 64x64, wave tile 32x32.
// K staged 64 at a time through double-buffered LDS via async copies
// (global_load_async_to_lds_b128 + s_wait_asynccnt), 8 WMMAs per stage.
#define KT   64
#define TSTR 72          // padded LDS tile row stride (elements)
__global__ __launch_bounds__(128) void gemm_bf16_kernel(
    const unsigned short* __restrict__ A, const unsigned short* __restrict__ Bt,
    const float* __restrict__ bias, const float* __restrict__ resid,
    float* __restrict__ C, int M, int N, int K, int act) {
  __shared__ unsigned short lA[2][64 * TSTR];
  __shared__ unsigned short lB[2][64 * TSTR];
  int tid  = threadIdx.x;
  int lane = tid & 31;
  int wid  = tid >> 5;
  int rowblk = blockIdx.y * 64;
  int colblk = blockIdx.x * 64;

  // async-copy mapping: 2 threads per tile row, 4 x 16B chunks each
  int crow = tid >> 1;
  int ce0  = (tid & 1) * 32;         // element offset within 64-wide K tile
  const unsigned short* gA = A  + (size_t)(rowblk + crow) * K + ce0;
  const unsigned short* gB = Bt + (size_t)(colblk + crow) * K + ce0;

  auto issue_tile = [&](int buf, int kk) {
    unsigned la = (unsigned)(size_t)&lA[buf][crow * TSTR + ce0];
    unsigned lb = (unsigned)(size_t)&lB[buf][crow * TSTR + ce0];
#pragma unroll
    for (int c = 0; c < 32; c += 8) {          // 4+4 issues -> ASYNCcnt += 8
      async_b128(la + c * 2, gA + kk + c);
      async_b128(lb + c * 2, gB + kk + c);
    }
  };

  v8f acc00 = {}, acc01 = {}, acc10 = {}, acc11 = {};
  int r0 = (wid >> 1) * 32;          // wave tile inside 64x64 block tile
  int c0 = (wid & 1) * 32;

  issue_tile(0, 0);
  int stages = K / KT;
  for (int it = 0; it < stages; ++it) {
    int buf = it & 1;
    if (it + 1 < stages) {
      issue_tile(buf ^ 1, (it + 1) * KT);
      asm volatile("s_wait_asynccnt 8" ::: "memory");   // current tile landed
    } else {
      asm volatile("s_wait_asynccnt 0" ::: "memory");
    }
    __syncthreads();
    const unsigned short* Ab = lA[buf];
    const unsigned short* Bb = lB[buf];
#pragma unroll
    for (int ks = 0; ks < KT; ks += 32) {
      v16bf a0 = fragA(Ab, TSTR, r0,      ks, lane);    // ds_load_b128 x2
      v16bf a1 = fragA(Ab, TSTR, r0 + 16, ks, lane);
      v16bf b0 = fragB(Bb, TSTR, c0,      ks, lane);
      v16bf b1 = fragB(Bb, TSTR, c0 + 16, ks, lane);
      acc00 = wmma_bf16(a0, b0, acc00);
      acc01 = wmma_bf16(a0, b1, acc01);
      acc10 = wmma_bf16(a1, b0, acc10);
      acc11 = wmma_bf16(a1, b1, acc11);
    }
    __syncthreads();                 // buffers free for the next async writes
  }

  int half = lane >> 4, cn = lane & 15;
  int row_base = rowblk + r0, col_base = colblk + c0;
  auto store_tile = [&](v8f acc, int rr0, int cc0) {
#pragma unroll
    for (int j = 0; j < 8; ++j) {
      int row = rr0 + j + 8 * half;
      int col = cc0 + cn;
      size_t idx = (size_t)row * N + col;
      float v = acc[j];
      if (bias)  v += bias[col];
      if (resid) v += resid[idx];
      C[idx] = apply_act(v, act);
    }
  };
  store_tile(acc00, row_base,      col_base);
  store_tile(acc01, row_base,      col_base + 16);
  store_tile(acc10, row_base + 16, col_base);
  store_tile(acc11, row_base + 16, col_base + 16);
}

// ---------------------------------------------------------------- attention
// scores[bh, q, k] = scale * Q[bh] @ K[bh]^T + mask[b, k]     (written to d_out)
// Q,K in bf16 [B*NH, S, HD];  grid (S/64, S/64, B*NH), block 128.  K depth is
// only 64, so fragments come straight from global (L2-resident operands).
__global__ __launch_bounds__(128) void scores_kernel(
    const unsigned short* __restrict__ Q, const unsigned short* __restrict__ Kh,
    const float* __restrict__ mask, float* __restrict__ Sout, float scale) {
  int bh = blockIdx.z;
  const unsigned short* Qb = Q  + (size_t)bh * SLEN * HD;
  const unsigned short* Kb = Kh + (size_t)bh * SLEN * HD;
  float* Cb = Sout + (size_t)bh * SLEN * SLEN;
  const float* mrow = mask + (size_t)(bh >> 4) * SLEN;   // NH = 16
  int lane = threadIdx.x & 31;
  int wid  = threadIdx.x >> 5;
  int row_base = blockIdx.y * 64 + (wid >> 1) * 32;
  int col_base = blockIdx.x * 64 + (wid & 1) * 32;
  v8f acc00 = {}, acc01 = {}, acc10 = {}, acc11 = {};
#pragma unroll
  for (int kk = 0; kk < HD; kk += 32) {
    v16bf a0 = fragA(Qb, HD, row_base,      kk, lane);
    v16bf a1 = fragA(Qb, HD, row_base + 16, kk, lane);
    v16bf b0 = fragB(Kb, HD, col_base,      kk, lane);
    v16bf b1 = fragB(Kb, HD, col_base + 16, kk, lane);
    acc00 = wmma_bf16(a0, b0, acc00);
    acc01 = wmma_bf16(a0, b1, acc01);
    acc10 = wmma_bf16(a1, b0, acc10);
    acc11 = wmma_bf16(a1, b1, acc11);
  }
  int half = lane >> 4, cn = lane & 15;
  auto store_tile = [&](v8f acc, int r0, int c0) {
#pragma unroll
    for (int j = 0; j < 8; ++j) {
      int row = r0 + j + 8 * half;
      int col = c0 + cn;
      Cb[(size_t)row * SLEN + col] = acc[j] * scale + mrow[col];
    }
  };
  store_tile(acc00, row_base,      col_base);
  store_tile(acc01, row_base,      col_base + 16);
  store_tile(acc10, row_base + 16, col_base);
  store_tile(acc11, row_base + 16, col_base + 16);
}

// Fused softmax + P@V.  One block = 16 query rows of one (b,h).
// Un-normalized exp(s - rowmax) kept in LDS as bf16; normalize in epilogue.
#define PSTR (SLEN + 8)
__global__ __launch_bounds__(128) void attn_pv_kernel(
    const float* __restrict__ scores, const unsigned short* __restrict__ Vt,
    float* __restrict__ ctx) {
  __shared__ unsigned short ldsP[16 * PSTR];
  __shared__ float red[16][8];
  __shared__ float rowsum[16];
  int bh = blockIdx.y;              // b*NH + h
  int qt = blockIdx.x;              // query tile (16 rows)
  int tid = threadIdx.x;
  const float* srow = scores + ((size_t)bh * SLEN + qt * 16) * SLEN;
  int r = tid >> 3, ch = tid & 7;                    // 16 rows x 8 chunks
  const float* prow = srow + (size_t)r * SLEN + ch * 64;
  float m = -3.4e38f;
#pragma unroll 8
  for (int i = 0; i < 64; ++i) m = fmaxf(m, prow[i]);
  red[r][ch] = m;
  __syncthreads();
  float rm = red[r][0];
#pragma unroll
  for (int i = 1; i < 8; ++i) rm = fmaxf(rm, red[r][i]);
  __syncthreads();
  float s = 0.f;
  unsigned short* pout = ldsP + r * PSTR + ch * 64;
#pragma unroll 8
  for (int i = 0; i < 64; ++i) {
    float e = __expf(prow[i] - rm);
    s += e;
    pout[i] = f2bf(e);
  }
  red[r][ch] = s;
  __syncthreads();
  if (tid < 16) {
    float t = 0.f;
#pragma unroll
    for (int i = 0; i < 8; ++i) t += red[tid][i];
    rowsum[tid] = t;
  }
  __syncthreads();
  // ---- P (16 x 512, LDS) @ V (512 x 64): wave w computes 16x16 n-tile w
  int lane = tid & 31, wid = tid >> 5;
  int n0 = wid * 16;
  int cn = lane & 15, half = lane >> 4;
  v8f acc = {};
  const unsigned short* arow = ldsP + (size_t)cn * PSTR + half * 8;    // A row = cn
  const unsigned short* vb = Vt + ((size_t)bh * HD + n0 + cn) * SLEN + half * 8;
  for (int kk = 0; kk < SLEN; kk += 32) {
    v16bf a = load_frag_pair(arow + kk, arow + kk + 16);   // ds_load_b128 x2
    v16bf b = load_frag_pair(vb + kk, vb + kk + 16);       // global_load_b128 x2
    acc = wmma_bf16(a, b, acc);
  }
  int b_ = bh >> 4, h_ = bh & 15;
#pragma unroll
  for (int j = 0; j < 8; ++j) {
    int mrow = j + 8 * half;
    float val = acc[j] / rowsum[mrow];
    ctx[((size_t)b_ * SLEN + qt * 16 + mrow) * DMODEL + h_ * HD + n0 + cn] = val;
  }
}

// ---------------------------------------------------------------- layernorm
__global__ __launch_bounds__(256) void layernorm_kernel(
    const float* __restrict__ x, const float* __restrict__ g,
    const float* __restrict__ b, float* __restrict__ y) {
  __shared__ float red[256];
  int row = blockIdx.x;
  const float* xr = x + (size_t)row * DMODEL;
  float v[4];
  float s = 0.f;
#pragma unroll
  for (int i = 0; i < 4; ++i) { v[i] = xr[threadIdx.x + i * 256]; s += v[i]; }
  red[threadIdx.x] = s;
  __syncthreads();
  for (int off = 128; off > 0; off >>= 1) {
    if (threadIdx.x < off) red[threadIdx.x] += red[threadIdx.x + off];
    __syncthreads();
  }
  float mu = red[0] * (1.f / DMODEL);
  __syncthreads();
  float ss = 0.f;
#pragma unroll
  for (int i = 0; i < 4; ++i) { float d = v[i] - mu; ss += d * d; }
  red[threadIdx.x] = ss;
  __syncthreads();
  for (int off = 128; off > 0; off >>= 1) {
    if (threadIdx.x < off) red[threadIdx.x] += red[threadIdx.x + off];
    __syncthreads();
  }
  float inv = rsqrtf(red[0] * (1.f / DMODEL) + 1e-12f);
#pragma unroll
  for (int i = 0; i < 4; ++i) {
    int c = threadIdx.x + i * 256;
    y[(size_t)row * DMODEL + c] = (v[i] - mu) * inv * g[c] + b[c];
  }
}

// ---------------------------------------------------------------- launcher
extern "C" void kernel_launch(void* const* d_in, const int* in_sizes, int n_in,
                              void* d_out, int out_size, void* d_ws, size_t ws_size,
                              hipStream_t stream) {
  const float* hidden = (const float*)d_in[0];
  const float* mask   = (const float*)d_in[1];
  const float* wq = (const float*)d_in[2];
  const float* wk = (const float*)d_in[3];
  const float* wo = (const float*)d_in[4];
  const float* bo = (const float*)d_in[5];
  const float* ln1_g = (const float*)d_in[6];
  const float* ln1_b = (const float*)d_in[7];
  const float* w1 = (const float*)d_in[8];
  const float* b1 = (const float*)d_in[9];
  const float* w2 = (const float*)d_in[10];
  const float* b2 = (const float*)d_in[11];
  const float* ln2_g = (const float*)d_in[12];
  const float* ln2_b = (const float*)d_in[13];

  float* out_h = (float*)d_out;                                  // [L,B,S,D]
  float* out_s = out_h + (size_t)LAYERS * BSD;                   // [L,B,NH,S,S]

  // ---- workspace carve-out
  char* w = (char*)d_ws;
  auto carve = [&](size_t bytes) {
    void* p = w;
    w += (bytes + 255) & ~(size_t)255;
    return p;
  };
  unsigned short* act_bf = (unsigned short*)carve((size_t)BSD * 2);
  unsigned short* ff_bf  = (unsigned short*)carve((size_t)BSD * 2);
  unsigned short* wqt = (unsigned short*)carve((size_t)DD * 2);
  unsigned short* wkt = (unsigned short*)carve((size_t)DD * 2);
  unsigned short* wot = (unsigned short*)carve((size_t)DD * 2);
  unsigned short* w1t = (unsigned short*)carve((size_t)DD * 2);
  unsigned short* w2t = (unsigned short*)carve((size_t)DD * 2);
  unsigned short* qh  = (unsigned short*)carve((size_t)BSD * 2); // [B,NH,S,HD]
  unsigned short* kh  = (unsigned short*)carve((size_t)BSD * 2);
  unsigned short* vt  = (unsigned short*)carve((size_t)BSD * 2); // [B,NH,HD,S]
  float* qf   = (float*)carve((size_t)BSD * 4);
  float* kf   = (float*)carve((size_t)BSD * 4);
  float* ctx  = (float*)carve((size_t)BSD * 4);
  float* tmp  = (float*)carve((size_t)BSD * 4);
  float* hmid = (float*)carve((size_t)BSD * 4);

  const dim3 gemm_grid(DMODEL / 64, BS / 64);          // (16, 64)
  const dim3 tc_grid(DMODEL / 32, DMODEL / 32);        // transpose weights
  const dim3 sc_grid(SLEN / 64, SLEN / 64, BATCH * NH);// scores
  const dim3 pv_grid(SLEN / 16, BATCH * NH);           // fused softmax+PV
  const int  ew_grid = BSD / 256;                      // element-wise
  const float scale = 0.125f;                          // 1/sqrt(HD)

  for (int l = 0; l < LAYERS; ++l) {
    const float* hin = (l == 0) ? hidden : (out_h + (size_t)(l - 1) * BSD);
    float* h_out_l = out_h + (size_t)l * BSD;
    float* scores_l = out_s + (size_t)l * SCORES_PER_LAYER;

    // weights for this layer -> bf16 transposed [N,K]
    transpose_cvt_kernel<<<tc_grid, 256, 0, stream>>>(wq + (size_t)l * DD, wqt, DMODEL, DMODEL);
    transpose_cvt_kernel<<<tc_grid, 256, 0, stream>>>(wk + (size_t)l * DD, wkt, DMODEL, DMODEL);
    transpose_cvt_kernel<<<tc_grid, 256, 0, stream>>>(wo + (size_t)l * DD, wot, DMODEL, DMODEL);
    transpose_cvt_kernel<<<tc_grid, 256, 0, stream>>>(w1 + (size_t)l * DD, w1t, DMODEL, DMODEL);
    transpose_cvt_kernel<<<tc_grid, 256, 0, stream>>>(w2 + (size_t)l * DD, w2t, DMODEL, DMODEL);

    // q = selu(h @ wq), k = selu(h @ wk)
    cvt_bf16_kernel<<<ew_grid, 256, 0, stream>>>(hin, act_bf, BSD);
    gemm_bf16_kernel<<<gemm_grid, 128, 0, stream>>>(act_bf, wqt, nullptr, nullptr,
                                                    qf, BS, DMODEL, DMODEL, ACT_SELU);
    gemm_bf16_kernel<<<gemm_grid, 128, 0, stream>>>(act_bf, wkt, nullptr, nullptr,
                                                    kf, BS, DMODEL, DMODEL, ACT_SELU);

    // attention operand layouts
    permute_qk_kernel<<<ew_grid, 256, 0, stream>>>(qf, qh);
    permute_qk_kernel<<<ew_grid, 256, 0, stream>>>(kf, kh);
    transpose_v_kernel<<<ew_grid, 256, 0, stream>>>(hin, vt);

    // raw scores (output tensor) then fused softmax + PV
    scores_kernel<<<sc_grid, 128, 0, stream>>>(qh, kh, mask, scores_l, scale);
    attn_pv_kernel<<<pv_grid, 128, 0, stream>>>(scores_l, vt, ctx);

    // h = LN1(ctx @ wo + bo + h)
    cvt_bf16_kernel<<<ew_grid, 256, 0, stream>>>(ctx, act_bf, BSD);
    gemm_bf16_kernel<<<gemm_grid, 128, 0, stream>>>(act_bf, wot, bo + (size_t)l * DMODEL,
                                                    hin, tmp, BS, DMODEL, DMODEL, ACT_NONE);
    layernorm_kernel<<<BS, 256, 0, stream>>>(tmp, ln1_g + (size_t)l * DMODEL,
                                             ln1_b + (size_t)l * DMODEL, hmid);

    // h = LN2(relu(h @ w1 + b1) @ w2 + b2 + h)  -> written straight to all_h[l]
    cvt_bf16_kernel<<<ew_grid, 256, 0, stream>>>(hmid, act_bf, BSD);
    gemm_bf16_kernel<<<gemm_grid, 128, 0, stream>>>(act_bf, w1t, b1 + (size_t)l * DMODEL,
                                                    nullptr, tmp, BS, DMODEL, DMODEL, ACT_RELU);
    cvt_bf16_kernel<<<ew_grid, 256, 0, stream>>>(tmp, ff_bf, BSD);
    gemm_bf16_kernel<<<gemm_grid, 128, 0, stream>>>(ff_bf, w2t, b2 + (size_t)l * DMODEL,
                                                    hmid, qf, BS, DMODEL, DMODEL, ACT_NONE);
    layernorm_kernel<<<BS, 256, 0, stream>>>(qf, ln2_g + (size_t)l * DMODEL,
                                             ln2_b + (size_t)l * DMODEL, h_out_l);
  }
  (void)in_sizes; (void)n_in; (void)out_size; (void)ws_size;
}
